// SimpleRNN_85555748536471
// MI455X (gfx1250) — matrix operand order repaired and checked by reading, not compiled, not run on hardware
//
#include <hip/hip_runtime.h>

typedef __bf16 v16bf __attribute__((ext_vector_type(16)));
typedef float  v8f   __attribute__((ext_vector_type(8)));

#define BB   64
#define II   32
#define JJ   32
#define HID  256
#define NB   2048          // batch per LSTM step (B*J == B*I)
#define FH   1024          // 4*HID
#define FC   1024

__device__ __forceinline__ float sigm(float x) { return 1.0f / (1.0f + __expf(-x)); }

// ---- fragment loaders -------------------------------------------------------
// A fragment (16x32 bf16) straight from a row-major bf16 matrix.
// ISA layout: lane m=lane&15; half=lane>>4; elems 0-7 = K kb+half*8..+7,
// elems 8-15 = K kb+16+half*8..+7  -> two 16B loads per lane.
__device__ __forceinline__ v16bf load_afrag(const __bf16* __restrict__ A,
                                            long row_base, int lda, int kb) {
  int lane = threadIdx.x & 31;
  int m = lane & 15, half = lane >> 4;
  const __bf16* p = A + (row_base + m) * (long)lda + kb + half * 8;
  union { uint4 u[2]; v16bf v; } t;
  t.u[0] = *(const uint4*)p;
  t.u[1] = *(const uint4*)(p + 16);
  return t.v;
}

// B fragment from an LDS-staged panel: 512 bf16 (1 KB) per fragment, 16/lane.
__device__ __forceinline__ v16bf lds_bfrag(const __bf16* sB, int f) {
  int lane = threadIdx.x & 31;
  return *(const v16bf*)(sB + (size_t)f * 512 + lane * 16);
}

// ---- weight packing ---------------------------------------------------------
// Pack f32 [Ksrc x Nn] into bf16 WMMA-B tiles (K padded with zeros to 32*ktiles).
// B layout (32x16 bf16): lane n=lane&15, half=lane>>4, elem e<8 -> K = half*8+e, e>=8 -> K = 16+half*8+(e-8).
__global__ void __launch_bounds__(256) pack_b_kernel(const float* __restrict__ W,
                                                     __bf16* __restrict__ P,
                                                     int Ksrc, int Nn, int total) {
  int idx = blockIdx.x * blockDim.x + threadIdx.x;
  if (idx >= total) return;
  int e    = idx & 15;
  int lane = (idx >> 4) & 31;
  int tile = idx >> 9;
  int ntiles = Nn >> 4;
  int nt = tile % ntiles;
  int kt = tile / ntiles;
  int nl = lane & 15, half = lane >> 4;
  int kl = (e < 8) ? (half * 8 + e) : (16 + half * 8 + (e - 8));
  int k = kt * 32 + kl;
  int n = nt * 16 + nl;
  float v = (k < Ksrc) ? W[(long)k * Nn + n] : 0.0f;
  P[idx] = (__bf16)v;
}

// ---- patch extraction: x[64,64,64,3] -> pV bf16 [I, 2048, 32] (K pad 12->32) --
__global__ void __launch_bounds__(256) patch_kernel(const float* __restrict__ x,
                                                    __bf16* __restrict__ pV) {
  int idx = blockIdx.x * blockDim.x + threadIdx.x;  // 32*2048*32
  if (idx >= II * NB * 32) return;
  int k = idx & 31;
  int n = (idx >> 5) & (NB - 1);
  int i = idx >> 16;
  float v = 0.0f;
  if (k < 12) {
    int pr = k / 6, pc = (k / 3) & 1, ch = k % 3;
    int b = n >> 5, j = n & 31;
    v = x[(((long)b * 64 + (2 * i + pr)) * 64 + (2 * j + pc)) * 3 + ch];
  }
  pV[idx] = (__bf16)v;
}

__global__ void __launch_bounds__(256) zero_hc_kernel(__bf16* __restrict__ h,
                                                      float* __restrict__ c, int n) {
  int i = blockIdx.x * blockDim.x + threadIdx.x;
  if (i < n) { h[i] = (__bf16)0.0f; c[i] = 0.0f; }
}

// ---- one LSTM timestep ------------------------------------------------------
// Block = 256 threads (8 waves). All waves share one hidden col-tile ct; each
// wave owns a different 16-row tile. The full B panel for ct (4 gates x
// (K2/32 input + 8 recurrent) k-tiles) is staged in LDS once, then each wave
// accumulates z = A2 @ Bk + h_prev @ Br + bias and runs the fused cell update.
__global__ void __launch_bounds__(256) lstm_step_kernel(
    const __bf16* __restrict__ A2, int lda2, int K2,       // input feats this step
    const __bf16* __restrict__ Bk,                         // packed input weights
    const __bf16* __restrict__ Br,                         // packed recurrent weights
    const float*  __restrict__ bias,                       // [1024]
    const __bf16* __restrict__ hin,                        // [NB x 256] bf16
    __bf16* __restrict__ hout,                             // [NB x 256] bf16
    float*  __restrict__ c,                                // [NB x 256] f32
    __bf16* __restrict__ outp,                             // scatter dest (bf16)
    int smod, int sdiv, int mod, int row_off, int col_off, int out_ld) {
  extern __shared__ __align__(16) char smem[];
  __bf16* sB = (__bf16*)smem;

  int wid = threadIdx.x >> 5;
  int ct = blockIdx.x & 15;              // hidden col tile (shared by block)
  int rt = (blockIdx.x >> 4) * 8 + wid;  // 0..127 row tile of 16 in NB

  // cooperative B-panel stage: input frags first, then recurrent frags
  int ktiles2 = K2 >> 5;
  int F = (ktiles2 + 8) * 4;             // total 1KB fragments
  {
    uint4* dst = (uint4*)smem;
    for (int idx = threadIdx.x; idx < F * 64; idx += 256) {
      int f = idx >> 6, off = idx & 63;
      const __bf16* src;
      if (f < ktiles2 * 4) {
        int kt = f >> 2, g = f & 3;
        src = Bk + ((size_t)(kt * 64 + g * 16 + ct) * 512);
      } else {
        int f2 = f - ktiles2 * 4;
        int kt = f2 >> 2, g = f2 & 3;
        src = Br + ((size_t)(kt * 64 + g * 16 + ct) * 512);
      }
      dst[idx] = ((const uint4*)src)[off];
    }
  }
  __syncthreads();

  v8f acc[4] = {v8f{}, v8f{}, v8f{}, v8f{}};

  // input contribution
  for (int kb = 0; kb < K2; kb += 32) {
    v16bf a = load_afrag(A2, (long)rt * 16, lda2, kb);
    int kt = kb >> 5;
#pragma unroll
    for (int g = 0; g < 4; ++g) {
      v16bf b = lds_bfrag(sB, kt * 4 + g);
      acc[g] = __builtin_amdgcn_wmma_f32_16x16x32_bf16(false, a, false, b,
                                                       (short)0, acc[g], false, false);
    }
  }
  // recurrent contribution (K = 256)
  for (int kb = 0; kb < HID; kb += 32) {
    v16bf a = load_afrag(hin, (long)rt * 16, HID, kb);
    int kt = kb >> 5;
#pragma unroll
    for (int g = 0; g < 4; ++g) {
      v16bf b = lds_bfrag(sB, (ktiles2 + kt) * 4 + g);
      acc[g] = __builtin_amdgcn_wmma_f32_16x16x32_bf16(false, a, false, b,
                                                       (short)0, acc[g], false, false);
    }
  }

  int lane = threadIdx.x & 31;
  int nl = lane & 15, half = lane >> 4;
  int Nc = ct * 16 + nl;
  float b0 = bias[Nc], b1 = bias[HID + Nc], b2 = bias[2 * HID + Nc], b3 = bias[3 * HID + Nc];
#pragma unroll
  for (int r = 0; r < 8; ++r) {
    int M = rt * 16 + r + 8 * half;            // batch row
    float zi = acc[0][r] + b0;
    float zf = acc[1][r] + b1;
    float zg = acc[2][r] + b2;
    float zo = acc[3][r] + b3;
    long ci = (long)M * HID + Nc;
    float cn = sigm(zf) * c[ci] + sigm(zi) * tanhf(zg);
    float hn = sigm(zo) * tanhf(cn);
    c[ci] = cn;
    hout[ci] = (__bf16)hn;
    long orow = (long)(M % mod) * smod + (long)(M / mod) * sdiv + row_off;
    outp[orow * (long)out_ld + col_off + Nc] = (__bf16)hn;
  }
}

// ---- dense + ReLU: out[65536x1024] = relu(A[65536x512] @ Wd + bd) -----------
// Block = 256 threads (8 waves) sharing one 64-col group; B panel (64 KB) in LDS.
__global__ void __launch_bounds__(256) dense_relu_kernel(
    const __bf16* __restrict__ A, const __bf16* __restrict__ Bp,
    const float* __restrict__ bias, float* __restrict__ out) {
  extern __shared__ __align__(16) char smem[];
  __bf16* sB = (__bf16*)smem;

  int wid = threadIdx.x >> 5;
  int cg = blockIdx.x & 15;               // group of 4 n-tiles (64 cols)
  int rt = (blockIdx.x >> 4) * 8 + wid;   // 0..4095

  {
    uint4* dst = (uint4*)smem;
    for (int idx = threadIdx.x; idx < 64 * 64; idx += 256) {  // 64 frags x 64 uint4
      int f = idx >> 6, off = idx & 63;
      int kt = f >> 2, g = f & 3;
      const __bf16* src = Bp + ((size_t)(kt * 64 + cg * 4 + g) * 512);
      dst[idx] = ((const uint4*)src)[off];
    }
  }
  __syncthreads();

  v8f acc[4] = {v8f{}, v8f{}, v8f{}, v8f{}};
  for (int kb = 0; kb < 512; kb += 32) {
    v16bf a = load_afrag(A, (long)rt * 16, 512, kb);
    int kt = kb >> 5;
#pragma unroll
    for (int g = 0; g < 4; ++g) {
      v16bf b = lds_bfrag(sB, kt * 4 + g);
      acc[g] = __builtin_amdgcn_wmma_f32_16x16x32_bf16(false, a, false, b,
                                                       (short)0, acc[g], false, false);
    }
  }
  int lane = threadIdx.x & 31;
  int nl = lane & 15, half = lane >> 4;
#pragma unroll
  for (int g = 0; g < 4; ++g) {
    int Nc = (cg * 4 + g) * 16 + nl;
    float bb = bias[Nc];
#pragma unroll
    for (int r = 0; r < 8; ++r) {
      int M = rt * 16 + r + 8 * half;
      float v = acc[g][r] + bb;
      out[(long)M * FC + Nc] = v > 0.0f ? v : 0.0f;
    }
  }
}

// ---- host driver ------------------------------------------------------------
extern "C" void kernel_launch(void* const* d_in, const int* in_sizes, int n_in,
                              void* d_out, int out_size, void* d_ws, size_t ws_size,
                              hipStream_t stream) {
  const float* x    = (const float*)d_in[0];
  const float* k_ud = (const float*)d_in[1];
  const float* r_ud = (const float*)d_in[2];
  const float* b_ud = (const float*)d_in[3];
  const float* k_du = (const float*)d_in[4];
  const float* r_du = (const float*)d_in[5];
  const float* b_du = (const float*)d_in[6];
  const float* k_lr = (const float*)d_in[7];
  const float* r_lr = (const float*)d_in[8];
  const float* b_lr = (const float*)d_in[9];
  const float* k_rl = (const float*)d_in[10];
  const float* r_rl = (const float*)d_in[11];
  const float* b_rl = (const float*)d_in[12];
  const float* Wd   = (const float*)d_in[13];
  const float* bd   = (const float*)d_in[14];
  float* out = (float*)d_out;

  char* ws = (char*)d_ws;
  auto alloc = [&](size_t bytes) {
    char* p = ws;
    ws += (bytes + 255) & ~(size_t)255;
    return p;
  };
  __bf16* pV    = (__bf16*)alloc((size_t)II * NB * 32 * 2);          // 4 MB
  __bf16* vseq  = (__bf16*)alloc((size_t)JJ * NB * 512 * 2);         // 64 MB [j][b][i][512]
  __bf16* hcat  = (__bf16*)alloc((size_t)BB * II * JJ * 512 * 2);    // 64 MB [(b*I+i)*J+j][512]
  __bf16* h0    = (__bf16*)alloc((size_t)NB * HID * 2);
  __bf16* h1    = (__bf16*)alloc((size_t)NB * HID * 2);
  float*  cbuf  = (float*) alloc((size_t)NB * HID * 4);
  __bf16* kud_p = (__bf16*)alloc((size_t)1  * 64 * 512 * 2);
  __bf16* kdu_p = (__bf16*)alloc((size_t)1  * 64 * 512 * 2);
  __bf16* rud_p = (__bf16*)alloc((size_t)8  * 64 * 512 * 2);
  __bf16* rdu_p = (__bf16*)alloc((size_t)8  * 64 * 512 * 2);
  __bf16* klr_p = (__bf16*)alloc((size_t)16 * 64 * 512 * 2);
  __bf16* krl_p = (__bf16*)alloc((size_t)16 * 64 * 512 * 2);
  __bf16* rlr_p = (__bf16*)alloc((size_t)8  * 64 * 512 * 2);
  __bf16* rrl_p = (__bf16*)alloc((size_t)8  * 64 * 512 * 2);
  __bf16* wd_p  = (__bf16*)alloc((size_t)16 * 64 * 512 * 2);

  auto pack = [&](const float* W, __bf16* P, int Ksrc, int Nn) {
    int ktiles = (Ksrc + 31) / 32;
    int total = ktiles * (Nn / 16) * 512;
    pack_b_kernel<<<(total + 255) / 256, 256, 0, stream>>>(W, P, Ksrc, Nn, total);
  };
  pack(k_ud, kud_p, 12, FH);
  pack(k_du, kdu_p, 12, FH);
  pack(r_ud, rud_p, HID, FH);
  pack(r_du, rdu_p, HID, FH);
  pack(k_lr, klr_p, 2 * HID, FH);
  pack(k_rl, krl_p, 2 * HID, FH);
  pack(r_lr, rlr_p, HID, FH);
  pack(r_rl, rrl_p, HID, FH);
  pack(Wd,   wd_p,  2 * HID, FC);

  {
    int total = II * NB * 32;
    patch_kernel<<<(total + 255) / 256, 256, 0, stream>>>(x, pV);
  }

  const int HC = NB * HID;
  auto run_dir = [&](const __bf16* Aseq, int lda2, int K2,
                     const __bf16* Bk, const __bf16* Br, const float* bias,
                     __bf16* outp, bool backward,
                     int smod, int sdiv, int mod, int col_off) {
    zero_hc_kernel<<<(HC + 255) / 256, 256, 0, stream>>>(h0, cbuf, HC);
    size_t shmem = (size_t)((K2 >> 5) + 8) * 4 * 1024;   // B panel bytes
    for (int s = 0; s < 32; ++s) {
      int t = backward ? (31 - s) : s;
      const __bf16* hin = (s & 1) ? h1 : h0;
      __bf16* hout      = (s & 1) ? h0 : h1;
      lstm_step_kernel<<<256, 256, shmem, stream>>>(
          Aseq + (size_t)t * NB * lda2, lda2, K2, Bk, Br, bias,
          hin, hout, cbuf, outp, smod, sdiv, mod, t, col_off, 512);
    }
  };

  // vertical sweeps: seq over i, batch n = b*J + j ; scatter to vseq[j][b][i][:]
  run_dir(pV, 32, 32, kud_p, rud_p, b_ud, vseq, false, BB * II, II, JJ, 0);
  run_dir(pV, 32, 32, kdu_p, rdu_p, b_du, vseq, true,  BB * II, II, JJ, HID);
  // horizontal sweeps: seq over j, batch n = b*I + i ; scatter to hcat[n*J+j][:]
  run_dir(vseq, 512, 512, klr_p, rlr_p, b_lr, hcat, false, JJ, 0, NB, 0);
  run_dir(vseq, 512, 512, krl_p, rrl_p, b_rl, hcat, true,  JJ, 0, NB, HID);

  // dense + relu -> d_out [65536 x 1024] f32
  dense_relu_kernel<<<8192, 256, 64 * 1024, stream>>>(hcat, wd_p, bd, out);
}
